// SelfAttention_56813827391502
// MI455X (gfx1250) — matrix operand order repaired and checked by reading, not compiled, not run on hardware
//
#include <hip/hip_runtime.h>
#include <hip/hip_bf16.h>

// ---------------------------------------------------------------------------
// Self-attention forward for B=2, L=2048, C=1024, H=16, D=64 on gfx1250.
// All matmul work (QKV proj, Q*K^T, P*V, out proj) runs on v_wmma_f32_16x16x32_f16
// with f32 accumulation; softmax is f32 with an online (flash) formulation.
// GEMM staging uses the CDNA5 async global->LDS path with a double-buffered
// pipeline synchronized via s_wait_asynccnt.
// ---------------------------------------------------------------------------

typedef __attribute__((ext_vector_type(16))) _Float16      v16h;
typedef __attribute__((ext_vector_type(8)))  float         v8f;

#define B_ 2
#define L_ 2048
#define C_ 1024
#define H_ 16
#define D_ 64

union FragA { v16h v; unsigned u[8]; };

// 16-bit A-matrix 16x32 layout (ISA 7.12.2): lane holds row M=lane&15,
// half hi=lane>>4; VGPR j holds K = 16*(j>>2) + 2*(j&3) + 8*hi (and K+1).
__device__ __forceinline__ int a_k0(int j, int hi) {
  return ((j >> 2) << 4) + ((j & 3) << 1) + (hi << 3);
}

// ---------------------------------------------------------------------------
// fp32 -> fp16 conversion (grid-stride)
// ---------------------------------------------------------------------------
__global__ void f32_to_f16(const float* __restrict__ in,
                           _Float16* __restrict__ out, int n) {
  int i = blockIdx.x * blockDim.x + threadIdx.x;
  int stride = gridDim.x * blockDim.x;
  for (; i < n; i += stride) out[i] = (_Float16)in[i];
}

// ---------------------------------------------------------------------------
// LDS-tiled WMMA GEMM: C[M,N] = A[M,K] * B[K,N], all row-major.
// Block: 128 threads (4 waves), 64x64 tile; wave -> 32x32 (2x2 WMMA frags).
// Double-buffered async global->LDS staging (ASYNCcnt pipeline).
// ---------------------------------------------------------------------------
template <typename OutT>
__global__ __launch_bounds__(128)
void gemm_wmma(const _Float16* __restrict__ A, const _Float16* __restrict__ Bm,
               OutT* __restrict__ Cm, int M, int N, int K) {
  // Padded strides: 40 halves (80B, rows stay 16B-aligned, conflict-free A frag
  // reads) and 72 halves (144B) for B.
  __shared__ __align__(16) _Float16 As[2][64][40];
  __shared__ __align__(16) _Float16 Bs[2][32][72];

  const int t    = threadIdx.x;
  const int lane = t & 31;
  const int wave = t >> 5;
  const int wm = wave >> 1, wn = wave & 1;
  const int hi = lane >> 4, l15 = lane & 15;
  const int rowBase = blockIdx.y * 64;
  const int colBase = blockIdx.x * 64;

  v8f c[2][2] = {};

  // Copy assignment: A tile 64x32 halves, B tile 32x64 halves; 16B per thread
  // per matrix per K-step via async DMA to LDS.
  const int ar = t >> 1, ahp = t & 1;   // A: 64 rows x 2 chunks of 16 halves
  const int br = t >> 2, bqp = t & 3;   // B: 32 rows x 4 chunks of 16 halves
  const _Float16* gA = &A[(size_t)(rowBase + ar) * K + ahp * 16];
  const _Float16* gB = &Bm[(size_t)br * N + colBase + bqp * 16];

  auto issue = [&](int k0, int buf) {
    unsigned la = (unsigned)(size_t)&As[buf][ar][ahp * 16];
    unsigned lb = (unsigned)(size_t)&Bs[buf][br][bqp * 16];
    const _Float16* pa = gA + k0;                 // 16B aligned
    const _Float16* pb = gB + (size_t)k0 * N;     // 16B aligned
    asm volatile("global_load_async_to_lds_b128 %0, %1, off"
                 :: "v"(la), "v"(pa) : "memory");
    asm volatile("global_load_async_to_lds_b128 %0, %1, off"
                 :: "v"(lb), "v"(pb) : "memory");
  };

  issue(0, 0);
  int buf = 0;
  for (int k0 = 0; k0 < K; k0 += 32, buf ^= 1) {
    if (k0 + 32 < K) {
      issue(k0 + 32, buf ^ 1);   // prefetch next tile into the other buffer
      // Async completions are in-order per wave: <=2 outstanding means the
      // older pair (tile k) has landed in LDS.
      asm volatile("s_wait_asynccnt 0x2" ::: "memory");
    } else {
      asm volatile("s_wait_asynccnt 0x0" ::: "memory");
    }
    __syncthreads();

    FragA a[2], b[2];
    for (int m = 0; m < 2; ++m) {
      const int r = wm * 32 + m * 16 + l15;
      for (int j = 0; j < 8; ++j)
        a[m].u[j] = *(const unsigned*)&As[buf][r][a_k0(j, hi)];
    }
    for (int n = 0; n < 2; ++n) {
      const int c0 = wn * 32 + n * 16;   // B frag: lane = N col, elems = K rows
      for (int j = 0; j < 8; ++j) {
        const int kk = a_k0(j, hi);
        b[n].v[2 * j]     = Bs[buf][kk][c0 + l15];
        b[n].v[2 * j + 1] = Bs[buf][kk + 1][c0 + l15];
      }
    }
    for (int m = 0; m < 2; ++m)
      for (int n = 0; n < 2; ++n)
        c[m][n] = __builtin_amdgcn_wmma_f32_16x16x32_f16(
            false, a[m].v, false, b[n].v, (short)0, c[m][n], false, false);
    __syncthreads();
  }

  for (int m = 0; m < 2; ++m)
    for (int n = 0; n < 2; ++n)
      for (int v = 0; v < 8; ++v) {
        const int r  = rowBase + wm * 32 + m * 16 + v + 8 * hi;
        const int cc = colBase + wn * 32 + n * 16 + l15;
        Cm[(size_t)r * N + cc] = (OutT)c[m][n][v];
      }
}

// ---------------------------------------------------------------------------
// RoPE + repack: qkv (B,L,3,H,D) f16 -> q (B,H,L,D), kT (B,H,D,L), v (B,H,L,D)
// One thread per (b,h,l, d-pair).
// ---------------------------------------------------------------------------
__global__ void rope_pack(const _Float16* __restrict__ qkv,
                          const float* __restrict__ fcos,
                          const float* __restrict__ fsin,
                          _Float16* __restrict__ qo,
                          _Float16* __restrict__ kto,
                          _Float16* __restrict__ vo) {
  const int i = blockIdx.x * blockDim.x + threadIdx.x;
  const int f = i & 31;                 // D/2 = 32 frequency pairs
  const int l = (i >> 5) & (L_ - 1);
  const int h = (i >> 16) & (H_ - 1);
  const int b = i >> 20;

  const size_t base = (size_t)(b * L_ + l) * (3 * C_);
  const int hd = h * D_ + 2 * f;
  const float cs = fcos[l * 32 + f];
  const float sn = fsin[l * 32 + f];

  const float q0 = (float)qkv[base + 0 * C_ + hd];
  const float q1 = (float)qkv[base + 0 * C_ + hd + 1];
  const float k0 = (float)qkv[base + 1 * C_ + hd];
  const float k1 = (float)qkv[base + 1 * C_ + hd + 1];
  const _Float16 v0 = qkv[base + 2 * C_ + hd];
  const _Float16 v1 = qkv[base + 2 * C_ + hd + 1];

  const size_t bh = (size_t)(b * H_ + h);
  const size_t qi = (bh * L_ + l) * D_ + 2 * f;
  qo[qi]     = (_Float16)(q0 * cs - q1 * sn);
  qo[qi + 1] = (_Float16)(q0 * sn + q1 * cs);
  vo[qi]     = v0;
  vo[qi + 1] = v1;
  const size_t ki = (bh * D_ + 2 * f) * L_ + l;   // transposed (d, key)
  kto[ki]      = (_Float16)(k0 * cs - k1 * sn);
  kto[ki + L_] = (_Float16)(k0 * sn + k1 * cs);
}

// ---------------------------------------------------------------------------
// Flash-style attention: one wave per (b,h, 16-query tile), 32-key tiles.
// S = Q K^T (4 WMMA), online softmax, O += P V (4 WMMA).
// ---------------------------------------------------------------------------
__global__ __launch_bounds__(128)
void attn_fwd(const _Float16* __restrict__ q, const _Float16* __restrict__ kt,
              const _Float16* __restrict__ v, _Float16* __restrict__ o) {
  __shared__ __align__(4) _Float16 P[4][16][34];   // per-wave P scratch (16x32, pad 2)

  const int lane = threadIdx.x & 31;
  const int wv   = threadIdx.x >> 5;
  const int wid  = blockIdx.x * 4 + wv;
  const int qt = wid & (L_ / 16 - 1);     // 128 query tiles
  const int bh = wid >> 7;
  const int b = bh >> 4, h = bh & 15;
  const int hi = lane >> 4, l15 = lane & 15;

  const _Float16* qb  = q  + (size_t)bh * L_ * D_;
  const _Float16* ktb = kt + (size_t)bh * D_ * L_;
  const _Float16* vb  = v  + (size_t)bh * L_ * D_;

  // Q A-fragments for d-chunks [0,32) and [32,64)
  FragA aq[2];
  for (int ch = 0; ch < 2; ++ch)
    for (int j = 0; j < 8; ++j)
      aq[ch].u[j] = *(const unsigned*)
          &qb[(size_t)(qt * 16 + l15) * D_ + ch * 32 + a_k0(j, hi)];

  v8f acc[4] = {};
  float mrow[8], lrow[8];
  for (int r = 0; r < 8; ++r) { mrow[r] = -3.0e38f; lrow[r] = 0.0f; }

  const float scale = 0.125f;   // 1/sqrt(64)

  for (int key0 = 0; key0 < L_; key0 += 32) {
    // ----- scores: two 16x16 C frags over this 32-key tile -----
    v8f s[2] = {};
    for (int ns = 0; ns < 2; ++ns) {
      for (int ch = 0; ch < 2; ++ch) {
        FragA bk;   // B frag from kT: lane = key col, elems = d rows
        for (int j = 0; j < 8; ++j) {
          const int kk = ch * 32 + a_k0(j, hi);
          bk.v[2 * j]     = ktb[(size_t)kk * L_ + key0 + ns * 16 + l15];
          bk.v[2 * j + 1] = ktb[(size_t)(kk + 1) * L_ + key0 + ns * 16 + l15];
        }
        s[ns] = __builtin_amdgcn_wmma_f32_16x16x32_f16(
            false, aq[ch].v, false, bk.v, (short)0, s[ns], false, false);
      }
    }

    // ----- online softmax update (rows live per half-wave, width-16 shuffles) --
    for (int r = 0; r < 8; ++r) {
      float s0 = s[0][r] * scale, s1 = s[1][r] * scale;
      float tmax = fmaxf(s0, s1);
      for (int off = 1; off < 16; off <<= 1)
        tmax = fmaxf(tmax, __shfl_xor(tmax, off, 16));
      const float mnew = fmaxf(mrow[r], tmax);
      const float corr = __expf(mrow[r] - mnew);
      mrow[r] = mnew;
      const float p0 = __expf(s0 - mnew);
      const float p1 = __expf(s1 - mnew);
      lrow[r] = lrow[r] * corr + p0 + p1;   // per-lane partial row sum
      for (int f = 0; f < 4; ++f) acc[f][r] *= corr;
      const int pr = r + 8 * hi;            // C-frag row -> LDS row
      P[wv][pr][l15]      = (_Float16)p0;
      P[wv][pr][16 + l15] = (_Float16)p1;
    }

    // P (16x32) back as an A fragment (same-wave DS ops are in-order)
    FragA pa;
    for (int j = 0; j < 8; ++j)
      pa.u[j] = *(const unsigned*)&P[wv][l15][a_k0(j, hi)];

    // ----- O += P * V : 4 WMMAs over d-chunks of 16 -----
    for (int f = 0; f < 4; ++f) {
      FragA bv;   // B frag from V: lane = d col, elems = key rows
      for (int j = 0; j < 8; ++j) {
        const int kk = key0 + a_k0(j, hi);
        bv.v[2 * j]     = vb[(size_t)kk * D_ + f * 16 + l15];
        bv.v[2 * j + 1] = vb[(size_t)(kk + 1) * D_ + f * 16 + l15];
      }
      acc[f] = __builtin_amdgcn_wmma_f32_16x16x32_f16(
          false, pa.v, false, bv.v, (short)0, acc[f], false, false);
    }
  }

  // Normalize and store to (B,L,H*D) f16 for the out-projection GEMM.
  for (int r = 0; r < 8; ++r) {
    float lr = lrow[r];
    for (int off = 1; off < 16; off <<= 1) lr += __shfl_xor(lr, off, 16);
    const float inv = 1.0f / lr;
    const int row_l = qt * 16 + r + 8 * hi;
    const size_t base = ((size_t)(b * L_ + row_l)) * C_ + h * D_;
    for (int f = 0; f < 4; ++f)
      o[base + f * 16 + l15] = (_Float16)(acc[f][r] * inv);
  }
}

// ---------------------------------------------------------------------------
// Host-side launch
// ---------------------------------------------------------------------------
extern "C" void kernel_launch(void* const* d_in, const int* in_sizes, int n_in,
                              void* d_out, int out_size, void* d_ws, size_t ws_size,
                              hipStream_t stream) {
  (void)in_sizes; (void)n_in; (void)out_size; (void)ws_size;

  const float* x    = (const float*)d_in[0];
  const float* fcos = (const float*)d_in[1];
  const float* fsin = (const float*)d_in[2];
  const float* wqkv = (const float*)d_in[3];
  const float* wout = (const float*)d_in[4];
  float* out = (float*)d_out;
  char* ws = (char*)d_ws;

  // Workspace layout (72 MiB total)
  _Float16* xh    = (_Float16*)(ws + ((size_t)0  << 20));  //  8 MiB: x f16
  _Float16* wqkvh = (_Float16*)(ws + ((size_t)8  << 20));  //  6 MiB: w_qkv f16
  _Float16* wouth = (_Float16*)(ws + ((size_t)14 << 20));  //  2 MiB: w_out f16
  _Float16* qkvh  = (_Float16*)(ws + ((size_t)16 << 20));  // 24 MiB: qkv f16
  _Float16* qbuf  = (_Float16*)(ws + ((size_t)40 << 20));  //  8 MiB: q (B,H,L,D)
  _Float16* ktbuf = (_Float16*)(ws + ((size_t)48 << 20));  //  8 MiB: k^T (B,H,D,L)
  _Float16* vbuf  = (_Float16*)(ws + ((size_t)56 << 20));  //  8 MiB: v (B,H,L,D)
  _Float16* attnh = (_Float16*)(ws + ((size_t)64 << 20));  //  8 MiB: attn out f16

  // 1) fp32 -> fp16 conversions
  f32_to_f16<<<1024, 256, 0, stream>>>(x,    xh,    B_ * L_ * C_);
  f32_to_f16<<<1024, 256, 0, stream>>>(wqkv, wqkvh, C_ * 3 * C_);
  f32_to_f16<<<1024, 256, 0, stream>>>(wout, wouth, C_ * C_);

  // 2) QKV projection: (4096 x 1024) x (1024 x 3072) -> f16
  gemm_wmma<_Float16><<<dim3(3 * C_ / 64, (B_ * L_) / 64), 128, 0, stream>>>(
      xh, wqkvh, qkvh, B_ * L_, 3 * C_, C_);

  // 3) RoPE + repack
  rope_pack<<<(B_ * H_ * L_ * (D_ / 2)) / 256, 256, 0, stream>>>(
      qkvh, fcos, fsin, qbuf, ktbuf, vbuf);

  // 4) Fused flash attention (one wave per (b,h,16-row query tile))
  attn_fwd<<<(B_ * H_ * (L_ / 16)) / 4, 128, 0, stream>>>(qbuf, ktbuf, vbuf, attnh);

  // 5) Output projection: (4096 x 1024) x (1024 x 1024) -> f32 d_out
  gemm_wmma<float><<<dim3(C_ / 64, (B_ * L_) / 64), 128, 0, stream>>>(
      attnh, wouth, out, B_ * L_, C_, C_);
}
